// CapsuleNetwork_34626026340529
// MI455X (gfx1250) — compile-verified, matrix operand order checked by reading, not run on hardware
//
#include <hip/hip_runtime.h>
#include <hip/hip_bf16.h>
#include <math.h>

// ---------------------------------------------------------------------------
// CapsuleNetwork forward for MI455X (gfx1250, wave32, WMMA, async-to-LDS).
//   conv2 (95% of FLOPs): implicit-GEMM, v_wmma_f32_16x16x32_bf16,
//                         double-buffered global_load_async_to_lds_b128
//   u_hat + decoder FCs : v_wmma_f32_16x16x4_f32 (pure fp32)
//   routing             : fused per-batch kernel, ~93KB dynamic LDS
// ---------------------------------------------------------------------------

typedef __attribute__((ext_vector_type(16))) __bf16 v16bf;
typedef __attribute__((ext_vector_type(8)))  __bf16 v8bf;
typedef __attribute__((ext_vector_type(8)))  float  v8f;
typedef __attribute__((ext_vector_type(2)))  float  v2f;

#define EPSQ 0.001f

// CDNA5 async copy: LDS[ldsaddr(lane)] <- MEM[gaddr(lane)], 16B, ASYNCcnt.
__device__ __forceinline__ void async_b128(const void* gaddr, const __bf16* lds) {
  unsigned loff = (unsigned)(uintptr_t)lds;   // low 32b of generic ptr = LDS offset
  asm volatile("global_load_async_to_lds_b128 %0, %1, off"
               :: "v"(loff), "v"(gaddr) : "memory");
}

// ---------------------------------------------------------------------------
// Conv1: [512,1,28,28] -> relu -> h bf16 channels-last [512,20,20,256]
// ---------------------------------------------------------------------------
__global__ __launch_bounds__(256) void conv1_relu_k(
    const float* __restrict__ x, const float* __restrict__ Wc,
    const float* __restrict__ bc, __bf16* __restrict__ h) {
  __shared__ float patch[81];
  const int pix = blockIdx.x;          // 0..399
  const int b   = blockIdx.y;          // 0..511
  const int oy = pix / 20, ox = pix % 20;
  const int t = threadIdx.x;           // output channel
  if (t < 81) {
    int ky = t / 9, kx = t % 9;
    patch[t] = x[b * 784 + (oy + ky) * 28 + (ox + kx)];
  }
  __syncthreads();
  float acc = bc[t];
  const float* wrow = Wc + t * 81;
#pragma unroll 9
  for (int k = 0; k < 81; ++k) acc += patch[k] * wrow[k];
  acc = fmaxf(acc, 0.f);
  h[(((b * 20 + oy) * 20) + ox) * 256 + t] = (__bf16)acc;
}

// ---------------------------------------------------------------------------
// Wp [256co,256ci,9,9] f32 -> Wpt bf16 [co][k], k = (ky*9+kx)*256 + ci.
// Row-major in k so conv2 B-tiles are contiguous 16B chunks (async-friendly).
// ---------------------------------------------------------------------------
__global__ void prep_wp_k(const float* __restrict__ Wp, __bf16* __restrict__ Wpt) {
  int f = blockIdx.x * 256 + threadIdx.x;
  if (f >= 256 * 20736) return;
  int co = f / 20736;
  int r  = f % 20736;
  int ci = r / 81;
  int pos = r % 81;
  Wpt[co * 20736 + pos * 256 + ci] = (__bf16)Wp[f];
}

// ---------------------------------------------------------------------------
// Conv2 implicit GEMM: M=18432 (b*36+oy*6+ox), N=256, K=20736, k-step 32.
// Block = 256 thr (8 waves, wm 0..3 x wn 0..1), wave tile 32x32 (2x2 WMMA),
// block tile 128x64. Double-buffered LDS fed by async b128 copies with the
// WMMA fragment swizzle folded into the LDS destination addresses:
//   chunk seg s (8 bf16 of K s*8..s*8+7) -> row*56 + (s&1)*16 + (s>>1)*8
// so a lane's fragment (m, kg=lane>>4) is 2 contiguous ds_load_b128 at
// row*56 + kg*16. Pipeline: wait(async<=3) -> barrier -> 4 WMMA -> barrier
// -> refill freed buffer for step+2.
// ---------------------------------------------------------------------------
__global__ __launch_bounds__(256) void conv2_wmma_k(
    const __bf16* __restrict__ h, const __bf16* __restrict__ Wpt,
    const float* __restrict__ bp, float* __restrict__ u) {
  __shared__ __bf16 As[2 * 128 * 56];
  __shared__ __bf16 Bs[2 * 64 * 56];
  const int pBase = blockIdx.x * 128;  // 144 blocks in M
  const int n0    = blockIdx.y * 64;   // 4 blocks in N
  const int t = threadIdx.x;
  const int wave = t >> 5, lane = t & 31;
  const int wm = wave >> 1, wn = wave & 1;

  // A loader: thread -> (row 0..127, seg pair {0,1} or {2,3})
  const int arow = t >> 1;
  const int as0  = (t & 1) * 2, as1 = as0 + 1;
  const int p    = pBase + arow;
  const int ab   = p / 36, arem = p % 36;
  const int aoy  = arem / 6, aox = arem % 6;
  const int adst0 = arow * 56 + (as0 & 1) * 16 + (as0 >> 1) * 8;
  const int adst1 = arow * 56 + (as1 & 1) * 16 + (as1 >> 1) * 8;
  // B loader: thread -> (n-in-block 0..63, seg 0..3)
  const int bnn = t >> 2, bseg = t & 3;
  const int bdst = bnn * 56 + (bseg & 1) * 16 + (bseg >> 1) * 8;
  const __bf16* brow = Wpt + (size_t)(n0 + bnn) * 20736 + bseg * 8;

  auto issue = [&](int kk, int bufi) {
    const int pos = kk >> 8, ci0 = kk & 255;
    const int ky = pos / 9, kx = pos % 9;
    const __bf16* ga =
        h + ((((ab * 20 + aoy * 2 + ky) * 20) + aox * 2 + kx) << 8) + ci0;
    __bf16* Ab = As + bufi * (128 * 56);
    __bf16* Bb = Bs + bufi * (64 * 56);
    async_b128(ga + as0 * 8, Ab + adst0);
    async_b128(ga + as1 * 8, Ab + adst1);
    async_b128(brow + kk, Bb + bdst);
  };

  const int kgrp = lane >> 4;
  const int am0 = wm * 32 + (lane & 15), am1 = am0 + 16;
  const int bn0 = wn * 32 + (lane & 15), bn1 = bn0 + 16;

  v8f acc00 = {0.f,0.f,0.f,0.f,0.f,0.f,0.f,0.f};
  v8f acc01 = acc00, acc10 = acc00, acc11 = acc00;

  issue(0, 0);
  issue(32, 1);
  int buf = 0;
  for (int k0 = 0; k0 < 20736; k0 += 32) {
    if (k0 + 32 < 20736) asm volatile("s_wait_asynccnt 0x3" ::: "memory");
    else                 asm volatile("s_wait_asynccnt 0x0" ::: "memory");
    __syncthreads();                         // current buffer complete everywhere

    const __bf16* Ab = As + buf * (128 * 56);
    const __bf16* Bb = Bs + buf * (64 * 56);
    v8bf al0 = *(const v8bf*)(Ab + am0 * 56 + kgrp * 16);
    v8bf ah0 = *(const v8bf*)(Ab + am0 * 56 + kgrp * 16 + 8);
    v8bf al1 = *(const v8bf*)(Ab + am1 * 56 + kgrp * 16);
    v8bf ah1 = *(const v8bf*)(Ab + am1 * 56 + kgrp * 16 + 8);
    v8bf bl0 = *(const v8bf*)(Bb + bn0 * 56 + kgrp * 16);
    v8bf bh0 = *(const v8bf*)(Bb + bn0 * 56 + kgrp * 16 + 8);
    v8bf bl1 = *(const v8bf*)(Bb + bn1 * 56 + kgrp * 16);
    v8bf bh1 = *(const v8bf*)(Bb + bn1 * 56 + kgrp * 16 + 8);
    v16bf a0, a1, b0, b1;
#pragma unroll
    for (int i = 0; i < 8; ++i) {
      a0[i] = al0[i]; a0[8 + i] = ah0[i];
      a1[i] = al1[i]; a1[8 + i] = ah1[i];
      b0[i] = bl0[i]; b0[8 + i] = bh0[i];
      b1[i] = bl1[i]; b1[8 + i] = bh1[i];
    }
    acc00 = __builtin_amdgcn_wmma_f32_16x16x32_bf16(false, a0, false, b0,
                                                    (short)0, acc00, false, false);
    acc01 = __builtin_amdgcn_wmma_f32_16x16x32_bf16(false, a0, false, b1,
                                                    (short)0, acc01, false, false);
    acc10 = __builtin_amdgcn_wmma_f32_16x16x32_bf16(false, a1, false, b0,
                                                    (short)0, acc10, false, false);
    acc11 = __builtin_amdgcn_wmma_f32_16x16x32_bf16(false, a1, false, b1,
                                                    (short)0, acc11, false, false);

    __syncthreads();                         // all waves done reading `buf`
    if (k0 + 64 < 20736) issue(k0 + 64, buf);
    buf ^= 1;
  }

  // epilogue: u[b][co][6][6] fp32 (+bias)
  const int coA = n0 + wn * 32 + (lane & 15);
  const int coB = coA + 16;
  const float biasA = bp[coA], biasB = bp[coB];
  const int mB0 = pBase + wm * 32 + (lane >> 4) * 8;
  const int mB1 = mB0 + 16;
#pragma unroll
  for (int v = 0; v < 8; ++v) {
    int p0 = mB0 + v, b0i = p0 / 36, r0 = p0 % 36;
    int p1 = mB1 + v, b1i = p1 / 36, r1 = p1 % 36;
    u[(b0i * 256 + coA) * 36 + r0] = acc00[v] + biasA;
    u[(b0i * 256 + coB) * 36 + r0] = acc01[v] + biasB;
    u[(b1i * 256 + coA) * 36 + r1] = acc10[v] + biasA;
    u[(b1i * 256 + coB) * 36 + r1] = acc11[v] + biasB;
  }
}

// ---------------------------------------------------------------------------
// squash along last-8 axis, in place (u_raw flat groups of 8 == capsules)
// ---------------------------------------------------------------------------
__global__ void squash8_k(float* __restrict__ u, int ncaps) {
  int i = blockIdx.x * 256 + threadIdx.x;
  if (i >= ncaps) return;
  float* p = u + i * 8;
  float v[8];
  float ns = 0.f;
#pragma unroll
  for (int o = 0; o < 8; ++o) { v[o] = p[o]; ns += v[o] * v[o]; }
  float sc = ns / (1.f + ns) / (EPSQ + sqrtf(ns));
#pragma unroll
  for (int o = 0; o < 8; ++o) p[o] = v[o] * sc;
}

// ---------------------------------------------------------------------------
// u_hat[b,c,jo] = sum_k Wd[c][jo][k]*u[b,c,k]; per-c GEMM M=512,N=160,K=8
// ---------------------------------------------------------------------------
__global__ __launch_bounds__(256) void uhat_wmma_k(
    const float* __restrict__ u, const float* __restrict__ Wd,
    float* __restrict__ uh) {
  const int c = blockIdx.x;
  const int wave = threadIdx.x >> 5, lane = threadIdx.x & 31;
  const int mt = blockIdx.y * 8 + wave;     // 0..31 (batch tiles)
  const int m  = mt * 16 + (lane & 15);
  const int kb = (lane >> 4) * 2;
  const float* urow = u + (m * 1152 + c) * 8;
  v2f a0 = *(const v2f*)(urow + kb);        // K 0..3
  v2f a1 = *(const v2f*)(urow + 4 + kb);    // K 4..7
  const float* wbase = Wd + c * 160 * 8;
  const int nidx = lane & 15;
  const int mB = mt * 16 + (lane >> 4) * 8;
  for (int nt = 0; nt < 10; ++nt) {
    const float* wrow = wbase + (nt * 16 + nidx) * 8;
    v2f b0 = *(const v2f*)(wrow + kb);
    v2f b1 = *(const v2f*)(wrow + 4 + kb);
    v8f acc = {0.f,0.f,0.f,0.f,0.f,0.f,0.f,0.f};
    acc = __builtin_amdgcn_wmma_f32_16x16x4_f32(false, a0, false, b0,
                                                (short)0, acc, false, false);
    acc = __builtin_amdgcn_wmma_f32_16x16x4_f32(false, a1, false, b1,
                                                (short)0, acc, false, false);
    const int jo = nt * 16 + nidx;
#pragma unroll
    for (int v = 0; v < 8; ++v)
      uh[((mB + v) * 1152 + c) * 160 + jo] = acc[v];
  }
}

// ---------------------------------------------------------------------------
// Fused dynamic routing (3 iters) + outputs. One block per batch element.
// ---------------------------------------------------------------------------
__global__ __launch_bounds__(256) void routing_k(
    const float* __restrict__ uh, const int* __restrict__ targets,
    float* __restrict__ out_v, float* __restrict__ out_idx,
    float* __restrict__ masked) {
  extern __shared__ float sm[];
  float* cij = sm;               // 11520
  float* bij = sm + 11520;       // 11520
  float* s   = sm + 23040;       // 160
  float* nsv = sm + 23200;       // 16
  float* scl = sm + 23216;       // 16
  const int b = blockIdx.x, t = threadIdx.x;
  const float* U = uh + (size_t)b * 184320;

  for (int i = t; i < 11520; i += 256) bij[i] = 0.f;
  __syncthreads();

  for (int it = 0; it < 3; ++it) {
    for (int c = t; c < 1152; c += 256) {     // c_ij = softmax_j(b_ij)
      const float* br = bij + c * 10;
      float mx = br[0];
      for (int j = 1; j < 10; ++j) mx = fmaxf(mx, br[j]);
      float e[10], sum = 0.f;
      for (int j = 0; j < 10; ++j) { e[j] = __expf(br[j] - mx); sum += e[j]; }
      float inv = 1.f / sum;
      float* cr = cij + c * 10;
      for (int j = 0; j < 10; ++j) cr[j] = e[j] * inv;
    }
    __syncthreads();
    if (t < 160) {                            // s[j,o], coalesced in c
      const int j = t >> 4;
      float acc = 0.f;
      for (int c = 0; c < 1152; ++c) acc += cij[c * 10 + j] * U[c * 160 + t];
      s[t] = acc;
    }
    __syncthreads();
    if (t < 10) {
      float ns = 0.f;
      for (int o = 0; o < 16; ++o) { float v = s[t * 16 + o]; ns += v * v; }
      nsv[t] = ns;
      scl[t] = ns / (1.f + ns) / (EPSQ + sqrtf(ns));
    }
    __syncthreads();
    if (it < 2) {                             // b_ij += <s_j, u_hat>
      for (int i = t; i < 11520; i += 256) {
        const int c = i / 10, j = i % 10;
        const float* ur = U + c * 160 + j * 16;
        const float* sr = s + j * 16;
        float d = 0.f;
#pragma unroll
        for (int o = 0; o < 16; ++o) d += sr[o] * ur[o];
        bij[i] += d;
      }
      __syncthreads();
    }
  }
  if (t < 160) {
    const int j = t >> 4;
    float v = scl[j] * s[t];
    out_v[b * 160 + t] = v;
    masked[b * 160 + t] = (j == targets[b]) ? v : 0.f;
  }
  if (t == 0) {
    int amax = 0; float best = nsv[0];
    for (int j = 1; j < 10; ++j) if (nsv[j] > best) { best = nsv[j]; amax = j; }
    out_idx[b] = (float)amax;
  }
}

// ---------------------------------------------------------------------------
// Generic FC: out = act(A[M,K] @ W[N,K]^T + bias), fp32 WMMA 16x16x4.
// act: 0=relu, 1=sigmoid. All dims multiples of 16 (784 = 49*16).
// ---------------------------------------------------------------------------
__global__ __launch_bounds__(256) void fc_wmma_k(
    const float* __restrict__ A, const float* __restrict__ W,
    const float* __restrict__ bias, float* __restrict__ out,
    int M, int K, int N, int act) {
  const int wave = threadIdx.x >> 5, lane = threadIdx.x & 31;
  const int tiles_n = N >> 4;
  const int tile = blockIdx.x * 8 + wave;
  if (tile >= (M >> 4) * tiles_n) return;
  const int mt = tile / tiles_n, nt = tile % tiles_n;
  const int m = mt * 16 + (lane & 15);
  const int n = nt * 16 + (lane & 15);
  const int kb = (lane >> 4) * 2;
  const float* arow = A + (size_t)m * K;
  const float* wrow = W + (size_t)n * K;
  v8f acc = {0.f,0.f,0.f,0.f,0.f,0.f,0.f,0.f};
  for (int k0 = 0; k0 < K; k0 += 4) {
    v2f a = *(const v2f*)(arow + k0 + kb);
    v2f w = *(const v2f*)(wrow + k0 + kb);
    acc = __builtin_amdgcn_wmma_f32_16x16x4_f32(false, a, false, w,
                                                (short)0, acc, false, false);
  }
  const float bv = bias[n];
  const int mBase = mt * 16 + (lane >> 4) * 8;
#pragma unroll
  for (int v = 0; v < 8; ++v) {
    float xv = acc[v] + bv;
    if (act == 0) xv = fmaxf(xv, 0.f);
    else          xv = 1.f / (1.f + __expf(-xv));
    out[(size_t)(mBase + v) * N + n] = xv;
  }
}

// ---------------------------------------------------------------------------
extern "C" void kernel_launch(void* const* d_in, const int* in_sizes, int n_in,
                              void* d_out, int out_size, void* d_ws, size_t ws_size,
                              hipStream_t stream) {
  const float* x  = (const float*)d_in[0];
  const int* tgt  = (const int*)d_in[1];
  const float* Wc = (const float*)d_in[2];
  const float* bc = (const float*)d_in[3];
  const float* Wp = (const float*)d_in[4];
  const float* bp = (const float*)d_in[5];
  const float* Wd = (const float*)d_in[6];
  const float* W1 = (const float*)d_in[7];
  const float* b1 = (const float*)d_in[8];
  const float* W2 = (const float*)d_in[9];
  const float* b2 = (const float*)d_in[10];
  const float* W3 = (const float*)d_in[11];
  const float* b3 = (const float*)d_in[12];
  float* out = (float*)d_out;   // [v 81920 | recon 401408 | index 512]

  char* ws = (char*)d_ws;
  __bf16* h   = (__bf16*)(ws);                  // 512*400*256 bf16 = 104857600 B
  __bf16* Wpt = (__bf16*)(ws + 104857600);      // 256*20736 bf16   = 10616832 B
  float*  u   = (float*)(ws + 115474432);       // 512*9216 f32     = 18874368 B
  float*  uh  = (float*)(ws + 134348800);       // 512*184320 f32   = 377487360 B
  float*  mk  = (float*)(ws + 511836160);       // 512*160 f32
  float*  z1  = (float*)(ws + 512163840);       // 512*512 f32
  float*  z2  = (float*)(ws + 513212416);       // 512*1024 f32

  conv1_relu_k<<<dim3(400, 512), 256, 0, stream>>>(x, Wc, bc, h);
  prep_wp_k<<<20736, 256, 0, stream>>>(Wp, Wpt);
  conv2_wmma_k<<<dim3(144, 4), 256, 0, stream>>>(h, Wpt, bp, u);
  squash8_k<<<(589824 + 255) / 256, 256, 0, stream>>>(u, 589824);
  uhat_wmma_k<<<dim3(1152, 4), 256, 0, stream>>>(u, Wd, uh);
  routing_k<<<512, 256, 23232 * sizeof(float), stream>>>(uh, tgt, out,
                                                         out + 483328, mk);
  fc_wmma_k<<<128, 256, 0, stream>>>(mk, W1, b1, z1, 512, 160, 512, 0);
  fc_wmma_k<<<256, 256, 0, stream>>>(z1, W2, b2, z2, 512, 512, 1024, 0);
  fc_wmma_k<<<196, 256, 0, stream>>>(z2, W3, b3, out + 81920, 512, 1024, 784, 1);
}
